// KNNAttention_5093831213089
// MI455X (gfx1250) — matrix operand order repaired and checked by reading, not compiled, not run on hardware
//
#include <hip/hip_runtime.h>
#include <hip/hip_bf16.h>
#include <math.h>
#include <stdint.h>

// ---------------------------------------------------------------------------
// KNN attention (PointTransformer-style) for MI455X / gfx1250, wave32.
//   GEMMs    : v_wmma_f32_16x16x32_bf16, bf16x2 (hi/lo) split ~= fp32 accuracy
//   A panels : pre-packed into wave32 fragment layout, staged into LDS with
//              global_load_async_to_lds_b128 (ASYNCcnt), read via ds_load
//   attention: one wave per point, L2-resident gathers, shuffle reductions
// ---------------------------------------------------------------------------

typedef __bf16 v16bf __attribute__((ext_vector_type(16)));
typedef float  v8f   __attribute__((ext_vector_type(8)));

#define NPTS   50000
#define CCH    128
#define KNN    16
#define KT     4          // K=128 in 4 chunks of 32
#define MT_PAD 3128       // 782 blocks * 4 row-tiles (>= 50000/16 = 3125)

#define AS3 __attribute__((address_space(3)))

__device__ __forceinline__ uint32_t lds_off32(const void* p) {
  return (uint32_t)(uintptr_t)(AS3 const void*)p;
}

// async DMA: 16 bytes global -> LDS, tracked by ASYNCcnt
__device__ __forceinline__ void async_ld_b128(uint32_t lds, const void* g) {
  asm volatile("global_load_async_to_lds_b128 %0, %1, off"
               :: "v"(lds), "v"(g) : "memory");
}
__device__ __forceinline__ void wait_async0() {
  asm volatile("s_wait_asynccnt 0x0" ::: "memory");
}

// ---------------------------------------------------------------------------
// Pack weights W (rows x 128, y = x @ W.T) into WMMA B-fragment layout:
// lane L holds column N = nt*16+(L&15), element j holds K like A (L<16: {0..7,16..23}, else {8..15,24..31}).
// ---------------------------------------------------------------------------
__global__ void pack_w_kernel(const float* __restrict__ W,
                              __bf16* __restrict__ ph, __bf16* __restrict__ pl,
                              int rows) {
  int t = blockIdx.x * blockDim.x + threadIdx.x;
  int total = (rows / 16) * KT * 32;
  if (t >= total) return;
  int lane = t & 31;
  int kt   = (t >> 5) & (KT - 1);
  int nt   = t >> 7;
  int k    = kt * 32 + ((lane & 16) ? 8 : 0);
  v16bf hv, lv;
#pragma unroll
  for (int j = 0; j < 16; ++j) {
    float w  = W[(size_t)(nt * 16 + (lane & 15)) * CCH + k + (j < 8 ? j : j + 8)];
    __bf16 h = (__bf16)w;
    hv[j] = h;
    lv[j] = (__bf16)(w - (float)h);
  }
  *(v16bf*)(ph + (size_t)t * 16) = hv;
  *(v16bf*)(pl + (size_t)t * 16) = lv;
}

// ---------------------------------------------------------------------------
// Pack activations A (nrows x 128 fp32) into WMMA A-fragment layout
// (16-bit A 16x32): lane<16 -> K {0..7,16..23}, lane>=16 -> K {8..15,24..31},
// lane&15 = row. Rows >= nrows are zero-filled (GEMM needs no load guards).
// ---------------------------------------------------------------------------
__global__ void pack_a_kernel(const float* __restrict__ A,
                              __bf16* __restrict__ ph, __bf16* __restrict__ pl,
                              int nrows, int mt_total) {
  int t = blockIdx.x * blockDim.x + threadIdx.x;
  int total = mt_total * KT * 32;
  if (t >= total) return;
  int lane = t & 31;
  int kt   = (t >> 5) & (KT - 1);
  int mt   = t >> 7;
  int row  = mt * 16 + (lane & 15);
  int koff = kt * 32 + ((lane & 16) ? 8 : 0);
  float a[16];
  if (row < nrows) {
    const float* p = A + (size_t)row * CCH + koff;
    const float4 f0 = *(const float4*)(p);
    const float4 f1 = *(const float4*)(p + 4);
    const float4 f2 = *(const float4*)(p + 16);
    const float4 f3 = *(const float4*)(p + 20);
    a[0]=f0.x;  a[1]=f0.y;  a[2]=f0.z;  a[3]=f0.w;
    a[4]=f1.x;  a[5]=f1.y;  a[6]=f1.z;  a[7]=f1.w;
    a[8]=f2.x;  a[9]=f2.y;  a[10]=f2.z; a[11]=f2.w;
    a[12]=f3.x; a[13]=f3.y; a[14]=f3.z; a[15]=f3.w;
  } else {
#pragma unroll
    for (int j = 0; j < 16; ++j) a[j] = 0.f;
  }
  v16bf hv, lv;
#pragma unroll
  for (int j = 0; j < 16; ++j) {
    __bf16 h = (__bf16)a[j];
    hv[j] = h;
    lv[j] = (__bf16)(a[j] - (float)h);
  }
  *(v16bf*)(ph + (size_t)t * 16) = hv;
  *(v16bf*)(pl + (size_t)t * 16) = lv;
}

// ---------------------------------------------------------------------------
// GEMM on packed fragments. Workgroup = 8 waves, 64 rows x (8*16) cols.
// A panel (16 fragments x 1KB, hi+lo = 32KB) is async-DMA'd into LDS once,
// shared by all 8 waves. Inner loop: ds_load fragments + 3x wmma (bf16x2).
// ---------------------------------------------------------------------------
__global__ __launch_bounds__(256) void gemm_packed_kernel(
    const __bf16* __restrict__ Ah, const __bf16* __restrict__ Al,
    const __bf16* __restrict__ Wh, const __bf16* __restrict__ Wl,
    float* __restrict__ OUT, const float* __restrict__ bias,
    int nrows, int ldo) {
  __shared__ __bf16 smem[2 * 16 * 512];   // [hi|lo][16 frags][512 bf16] = 32KB
  const int tid   = threadIdx.x;
  const int lane  = tid & 31;
  const int wave  = tid >> 5;
  const int ntile = blockIdx.y * 8 + wave;
  const int mt0   = blockIdx.x * 4;       // first 16-row tile of this block

  // ---- cooperative async copy: 16KB hi + 16KB lo of packed A fragments ----
  {
    const char* srcH = (const char*)(Ah + (size_t)mt0 * KT * 512);
    const char* srcL = (const char*)(Al + (size_t)mt0 * KT * 512);
    const uint32_t dstH = lds_off32(&smem[0]);
    const uint32_t dstL = lds_off32(&smem[16 * 512]);
#pragma unroll
    for (int i = 0; i < 4; ++i) {
      int c = (tid + i * 256) * 16;       // 1024 chunks of 16B per half
      async_ld_b128(dstH + c, srcH + c);
      async_ld_b128(dstL + c, srcL + c);
    }
  }
  wait_async0();
  __syncthreads();

  v8f acc[4];
#pragma unroll
  for (int f = 0; f < 4; ++f)
#pragma unroll
    for (int v = 0; v < 8; ++v) acc[f][v] = 0.f;

#pragma unroll
  for (int kt = 0; kt < KT; ++kt) {
    size_t wfrag = ((size_t)(ntile * KT + kt) * 32 + lane) * 16;
    v16bf bh = *(const v16bf*)(Wh + wfrag);
    v16bf bl = *(const v16bf*)(Wl + wfrag);
#pragma unroll
    for (int f = 0; f < 4; ++f) {
      const __bf16* ap = &smem[((size_t)(f * KT + kt) * 32 + lane) * 16];
      v16bf ah = *(const v16bf*)ap;
      v16bf al = *(const v16bf*)(ap + 16 * 512);
      // bf16x2: hi*hi + hi*lo + lo*hi, f32 accumulate
      acc[f] = __builtin_amdgcn_wmma_f32_16x16x32_bf16(false, ah, false, bh,
                                                       (short)0, acc[f], false, false);
      acc[f] = __builtin_amdgcn_wmma_f32_16x16x32_bf16(false, ah, false, bl,
                                                       (short)0, acc[f], false, false);
      acc[f] = __builtin_amdgcn_wmma_f32_16x16x32_bf16(false, al, false, bh,
                                                       (short)0, acc[f], false, false);
    }
  }

  const int  ncol = ntile * 16;
  const int  colw = ncol + (lane & 15);
  const float bv  = bias ? bias[colw] : 0.f;
  // C/D layout: lane<16 -> M = vgpr, N = lane; lane>=16 -> M = vgpr+8
#pragma unroll
  for (int f = 0; f < 4; ++f) {
#pragma unroll
    for (int v = 0; v < 8; ++v) {
      int r = mt0 * 16 + f * 16 + ((lane & 16) ? v + 8 : v);
      if (r < nrows) OUT[(size_t)r * ldo + colw] = acc[f][v] + bv;
    }
  }
}

// ---------------------------------------------------------------------------
// Attention: one wave per point, lane owns 4 channels. k/v gathers hit the
// 192MB L2 (k+v = 51MB resident).
// ---------------------------------------------------------------------------
__global__ __launch_bounds__(256) void knn_attn_kernel(
    const float* __restrict__ qkv, const int* __restrict__ knn,
    float* __restrict__ outa, int n) {
  int gwave = (int)((blockIdx.x * 256u + threadIdx.x) >> 5);
  int lane  = threadIdx.x & 31;
  if (gwave >= n) return;

  const float* qr = qkv + (size_t)gwave * (3 * CCH);
  float4 qv = *(const float4*)(qr + lane * 4);

  int idx[KNN];
#pragma unroll
  for (int j = 0; j < KNN; ++j) idx[j] = knn[gwave * KNN + j];

  float score = 0.f;
#pragma unroll
  for (int j = 0; j < KNN; ++j) {
    const float* kr = qkv + (size_t)idx[j] * (3 * CCH) + CCH;
    float4 kv = *(const float4*)(kr + lane * 4);
    float d = qv.x * kv.x + qv.y * kv.y + qv.z * kv.z + qv.w * kv.w;
#pragma unroll
    for (int off = 16; off > 0; off >>= 1) d += __shfl_xor(d, off, 32);
    if (lane == j) score = d;
  }
  score *= 0.08838834764831845f;                 // 128^-0.5

  float sm = (lane < KNN) ? score : -INFINITY;
  float m = sm;
#pragma unroll
  for (int off = 16; off > 0; off >>= 1) m = fmaxf(m, __shfl_xor(m, off, 32));
  float e = (lane < KNN) ? __expf(sm - m) : 0.f;
  float s = e;
#pragma unroll
  for (int off = 16; off > 0; off >>= 1) s += __shfl_xor(s, off, 32);
  float w = e / s;

  float4 av = {0.f, 0.f, 0.f, 0.f};
#pragma unroll
  for (int j = 0; j < KNN; ++j) {
    float wj = __shfl(w, j, 32);
    const float* vr = qkv + (size_t)idx[j] * (3 * CCH) + 2 * CCH;
    float4 vv = *(const float4*)(vr + lane * 4);
    av.x += wj * vv.x; av.y += wj * vv.y; av.z += wj * vv.z; av.w += wj * vv.w;
  }
  *(float4*)(outa + (size_t)gwave * CCH + lane * 4) = av;
}

// ---------------------------------------------------------------------------
extern "C" void kernel_launch(void* const* d_in, const int* in_sizes, int n_in,
                              void* d_out, int out_size, void* d_ws, size_t ws_size,
                              hipStream_t stream) {
  // inputs: 0:p 1:x 2:o 3:knn_idx 4:w_qkv 5:w_proj 6:b_proj
  const float* x      = (const float*)d_in[1];
  const int*   knn    = (const int*)d_in[3];
  const float* w_qkv  = (const float*)d_in[4];
  const float* w_proj = (const float*)d_in[5];
  const float* b_proj = (const float*)d_in[6];
  float* out = (float*)d_out;

  char* ws = (char*)d_ws;
  size_t off = 0;
  auto wsalloc = [&](size_t bytes) -> void* {
    void* p = ws + off;
    off = (off + bytes + 255) & ~(size_t)255;
    return p;
  };
  const size_t afrag_elems = (size_t)MT_PAD * KT * 512;   // per hi / lo
  float*  qkv  = (float*)wsalloc((size_t)NPTS * 3 * CCH * sizeof(float));   // 76.8 MB
  float*  attn = (float*)wsalloc((size_t)NPTS * CCH * sizeof(float));       // 25.6 MB
  __bf16* pxh  = (__bf16*)wsalloc(afrag_elems * sizeof(__bf16));            // 12.8 MB
  __bf16* pxl  = (__bf16*)wsalloc(afrag_elems * sizeof(__bf16));
  __bf16* pah  = (__bf16*)wsalloc(afrag_elems * sizeof(__bf16));
  __bf16* pal  = (__bf16*)wsalloc(afrag_elems * sizeof(__bf16));
  __bf16* pw1h = (__bf16*)wsalloc((size_t)24 * KT * 512 * sizeof(__bf16));
  __bf16* pw1l = (__bf16*)wsalloc((size_t)24 * KT * 512 * sizeof(__bf16));
  __bf16* pw2h = (__bf16*)wsalloc((size_t)8 * KT * 512 * sizeof(__bf16));
  __bf16* pw2l = (__bf16*)wsalloc((size_t)8 * KT * 512 * sizeof(__bf16));

  // 1) pack weights + activations into WMMA fragment layouts (bf16 hi/lo)
  pack_w_kernel<<<dim3((24 * KT * 32 + 255) / 256), 256, 0, stream>>>(w_qkv, pw1h, pw1l, 3 * CCH);
  pack_w_kernel<<<dim3((8 * KT * 32 + 255) / 256), 256, 0, stream>>>(w_proj, pw2h, pw2l, CCH);
  pack_a_kernel<<<dim3((MT_PAD * KT * 32 + 255) / 256), 256, 0, stream>>>(x, pxh, pxl, NPTS, MT_PAD);

  // 2) qkv = x @ w_qkv.T   (50000 x 384)
  gemm_packed_kernel<<<dim3(MT_PAD / 4, 3), 256, 0, stream>>>(
      pxh, pxl, pw1h, pw1l, qkv, nullptr, NPTS, 3 * CCH);

  // 3) gather + softmax attention over K=16 neighbors
  knn_attn_kernel<<<dim3((NPTS * 32 + 255) / 256), 256, 0, stream>>>(qkv, knn, attn, NPTS);

  // 4) pack attention output, then out = attn @ w_proj.T + b_proj
  pack_a_kernel<<<dim3((MT_PAD * KT * 32 + 255) / 256), 256, 0, stream>>>(attn, pah, pal, NPTS, MT_PAD);
  gemm_packed_kernel<<<dim3(MT_PAD / 4, 1), 256, 0, stream>>>(
      pah, pal, pw2h, pw2l, out, b_proj, NPTS, CCH);
}
